// MLA_37383395344545
// MI455X (gfx1250) — matrix operand order repaired and checked by reading, not compiled, not run on hardware
//
#include <hip/hip_runtime.h>

typedef __attribute__((ext_vector_type(16))) _Float16 v16h;
typedef __attribute__((ext_vector_type(8)))  _Float16 v8h;
typedef __attribute__((ext_vector_type(8)))  float    v8f;

#define DM   1024
#define SEQ  2048
#define NB   4
#define NH   16
#define DL   64
#define ROWS (NB * SEQ)   // 8192

static __device__ __forceinline__ v16h cat16(v8h lo, v8h hi) {
  return __builtin_shufflevector(lo, hi, 0,1,2,3,4,5,6,7,8,9,10,11,12,13,14,15);
}

static __device__ __forceinline__ v8f wmma_f16(v16h a, v16h b, v8f c) {
  return __builtin_amdgcn_wmma_f32_16x16x32_f16(
      /*neg_a=*/false, a, /*neg_b=*/false, b,
      /*c_mod=*/(short)0, c, /*reuse_a=*/false, /*reuse_b=*/false);
}

// A-operand tile (16x32 f16): k = (e>>3)*16 + half*8 + (e&7) -> two b128 loads
static __device__ __forceinline__ v16h loadA(const _Float16* rowp, int half) {
  return cat16(*(const v8h*)(rowp + half * 8), *(const v8h*)(rowp + 16 + half * 8));
}
// B-operand tile (32x16 f16) from K-contiguous storage: k = (e>>3)*16 + half*8 + (e&7)
static __device__ __forceinline__ v16h loadB(const _Float16* colp, int half) {
  return cat16(*(const v8h*)(colp + half * 8), *(const v8h*)(colp + 16 + half * 8));
}

// 16-lane butterfly max using DPP16 (no LDS, no dscnt waits).
template <int CTRL>
static __device__ __forceinline__ float dpp_max_step(float x) {
  const int y = __builtin_amdgcn_update_dpp(0, __float_as_int(x), CTRL, 0xF, 0xF, true);
  return fmaxf(x, __int_as_float(y));
}
static __device__ __forceinline__ float row_max16(float x) {
  x = dpp_max_step<0xB1>(x);   // quad_perm [1,0,3,2]  (xor 1)
  x = dpp_max_step<0x4E>(x);   // quad_perm [2,3,0,1]  (xor 2)
  x = dpp_max_step<0x141>(x);  // row_half_mirror      (pairs across 4s)
  x = dpp_max_step<0x140>(x);  // row_mirror           (pairs across 8s)
  return x;
}

// -------------------------------------------------------------------------
// Kernel 1: LayerNorm(x) -> xn16 (fp16), and cast raw x -> x16 (fp16).
// -------------------------------------------------------------------------
__global__ void ln_cast_kernel(const float* __restrict__ x,
                               const float* __restrict__ g,
                               const float* __restrict__ be,
                               _Float16* __restrict__ x16,
                               _Float16* __restrict__ xn16) {
  const int row = blockIdx.x;
  const float* xr = x + (size_t)row * DM;
  float v[4];
  float s = 0.f, s2 = 0.f;
  #pragma unroll
  for (int i = 0; i < 4; ++i) {
    v[i] = xr[threadIdx.x + 256 * i];
    s += v[i];
    s2 += v[i] * v[i];
  }
  #pragma unroll
  for (int off = 16; off > 0; off >>= 1) {
    s  += __shfl_xor(s,  off, 32);
    s2 += __shfl_xor(s2, off, 32);
  }
  __shared__ float red0[8], red1[8];
  const int wid = threadIdx.x >> 5;
  if ((threadIdx.x & 31) == 0) { red0[wid] = s; red1[wid] = s2; }
  __syncthreads();
  if (threadIdx.x == 0) {
    float ts = 0.f, ts2 = 0.f;
    #pragma unroll
    for (int i = 0; i < 8; ++i) { ts += red0[i]; ts2 += red1[i]; }
    red0[0] = ts * (1.0f / DM);
    red1[0] = ts2 * (1.0f / DM);
  }
  __syncthreads();
  const float mu  = red0[0];
  const float var = red1[0] - mu * mu;
  const float inv = rsqrtf(var + 1e-5f);
  #pragma unroll
  for (int i = 0; i < 4; ++i) {
    const int c = threadIdx.x + 256 * i;
    const float xn = (v[i] - mu) * inv * g[c] + be[c];
    x16 [(size_t)row * DM + c] = (_Float16)v[i];
    xn16[(size_t)row * DM + c] = (_Float16)xn;
  }
}

// -------------------------------------------------------------------------
// Kernel 2: cast fp32 weight W[K][N] -> fp16 transposed WT[N][K].
// -------------------------------------------------------------------------
__global__ void cast_wT_kernel(const float* __restrict__ w,
                               _Float16* __restrict__ wt) {
  const int idx = blockIdx.x * 256 + threadIdx.x;
  const int k = idx >> 10, n = idx & (DM - 1);
  wt[(size_t)n * DM + k] = (_Float16)w[idx];
}

// -------------------------------------------------------------------------
// Kernel 3: WMMA GEMM  C[8192][1024] = A @ WT^T + bias.  One wave computes a
// 32x64 tile, register double-buffered operands.  grid = (256, 16), block=32.
// MODE: 0 = f16 row-major out, 1 = f16 transposed out (out[n][m]), 2 = f32 out.
// __launch_bounds__(32,1): full VGPR file, no spills.
// -------------------------------------------------------------------------
template <int MODE>
__global__ __launch_bounds__(32, 1)
void gemm_wmma_kernel(const _Float16* __restrict__ A,
                      const _Float16* __restrict__ WT,
                      const float* __restrict__ bias,
                      _Float16* __restrict__ out16,
                      float* __restrict__ out32) {
  const int lane = threadIdx.x & 31;
  const int mrow = lane & 15;
  const int half = lane >> 4;
  const int m0 = blockIdx.x * 32;
  const int n0 = blockIdx.y * 64;

  const _Float16* aRow0 = A  + (size_t)(m0 + mrow) * DM;
  const _Float16* aRow1 = A  + (size_t)(m0 + 16 + mrow) * DM;
  const _Float16* bCol  = WT + (size_t)(n0 + mrow) * DM;  // + j*16*DM per col grp

  v8f acc[8] = {};
  v16h aC[2], bC[4], aN[2], bN[4];

  aC[0] = loadA(aRow0, half);
  aC[1] = loadA(aRow1, half);
  #pragma unroll
  for (int j = 0; j < 4; ++j) bC[j] = loadB(bCol + (size_t)j * 16 * DM, half);

  #pragma unroll 2
  for (int k0 = 0; k0 < DM; k0 += 32) {
    const int k1 = k0 + 32;
    if (k1 < DM) {
      aN[0] = loadA(aRow0 + k1, half);
      aN[1] = loadA(aRow1 + k1, half);
      #pragma unroll
      for (int j = 0; j < 4; ++j)
        bN[j] = loadB(bCol + (size_t)j * 16 * DM + k1, half);
    }
    #pragma unroll
    for (int j = 0; j < 4; ++j) acc[j]     = wmma_f16(aC[0], bC[j], acc[j]);
    #pragma unroll
    for (int j = 0; j < 4; ++j) acc[4 + j] = wmma_f16(aC[1], bC[j], acc[4 + j]);
    aC[0] = aN[0]; aC[1] = aN[1];
    #pragma unroll
    for (int j = 0; j < 4; ++j) bC[j] = bN[j];
  }

  #pragma unroll
  for (int mt = 0; mt < 2; ++mt) {
    #pragma unroll
    for (int j = 0; j < 4; ++j) {
      const int n = n0 + j * 16 + mrow;
      const float bv = bias[n];
      #pragma unroll
      for (int r = 0; r < 8; ++r) {
        const int m = m0 + mt * 16 + r + 8 * half;
        const float val = acc[mt * 4 + j][r] + bv;
        if (MODE == 0)      out16[(size_t)m * DM + n]   = (_Float16)val;
        else if (MODE == 1) out16[(size_t)n * ROWS + m] = (_Float16)val;  // [n][m]
        else                out32[(size_t)m * DM + n]   = val;
      }
    }
  }
}

// -------------------------------------------------------------------------
// Kernel 4: flash attention, one wave per (batch, head, 16-row Q block).
// 64-token blocks: S = Q K^T (8 WMMAs), max via DPP16 butterfly, row-sums via
// ones-WMMA, P re-layout through LDS, O += P V (8 WMMAs, b128 loads from vT).
// grid = (SEQ/16, NH, NB), block = 32.  __launch_bounds__(32,1): no spills.
// -------------------------------------------------------------------------
__global__ __launch_bounds__(32, 1)
void attn_wmma_kernel(const _Float16* __restrict__ q16,
                      const _Float16* __restrict__ k16,
                      const _Float16* __restrict__ vT16,
                      _Float16* __restrict__ ao16) {
  const int lane = threadIdx.x & 31;
  const int mrow = lane & 15;
  const int half = lane >> 4;
  const int qb = blockIdx.x * 16;
  const int h  = blockIdx.y;
  const int b  = blockIdx.z;
  const size_t base  = (size_t)b * SEQ * DM + (size_t)h * DL;        // q/k/ao: +t*1024+d
  const size_t vbase = (size_t)h * DL * ROWS + (size_t)b * SEQ;      // vT: +d*8192+t

  // Q block 16x64, pre-scaled by 1/sqrt(64) (exact in f16)
  v16h qa[2];
  #pragma unroll
  for (int ks = 0; ks < 2; ++ks) {
    qa[ks] = loadA(q16 + base + (size_t)(qb + mrow) * DM + ks * 32, half);
    qa[ks] = qa[ks] * (_Float16)0.125f;
  }

  v16h onesB;
  #pragma unroll
  for (int e = 0; e < 16; ++e) onesB[e] = (_Float16)1.0f;

  v8f o[4] = {};
  float mrun[8], lrun[8];
  #pragma unroll
  for (int r = 0; r < 8; ++r) { mrun[r] = -3.0e38f; lrun[r] = 0.f; }

  __shared__ __align__(16) _Float16 pl[16 * 64];

  #pragma unroll 1
  for (int t0 = 0; t0 < SEQ; t0 += 64) {
    // prefetch next 64-token K / vT block (one 128B line per lane)
    if (t0 + 64 < SEQ) {
      __builtin_prefetch(k16 + base + (size_t)(t0 + 64 + 2 * lane) * DM, 0, 0);
      __builtin_prefetch(vT16 + vbase + (size_t)lane * ROWS + t0 + 64, 0, 0);
    }

    // --- scores: four 16x16 tiles over 64 tokens ---
    v8f s[4];
    #pragma unroll
    for (int jt = 0; jt < 4; ++jt) {
      v8f c = {};
      #pragma unroll
      for (int ks = 0; ks < 2; ++ks) {
        const v16h kb =
            loadB(k16 + base + (size_t)(t0 + jt * 16 + mrow) * DM + ks * 32, half);
        c = wmma_f16(qa[ks], kb, c);
      }
      s[jt] = c;
    }

    // --- online softmax: max via DPP butterfly, exp + store P to LDS.
    //     alpha is consumed inside this loop (o and lrun rescale) so it does
    //     not stay live across the WMMA phase.
    #pragma unroll
    for (int r = 0; r < 8; ++r) {
      float m = fmaxf(fmaxf(s[0][r], s[1][r]), fmaxf(s[2][r], s[3][r]));
      m = row_max16(m);
      const float mn = fmaxf(mrun[r], m);
      const float alpha = __expf(mrun[r] - mn);
      mrun[r] = mn;
      lrun[r] *= alpha;
      const int row = r + 8 * half;
      #pragma unroll
      for (int jt = 0; jt < 4; ++jt) {
        const float p = __expf(s[jt][r] - mn);
        pl[row * 64 + jt * 16 + mrow] = (_Float16)p;
      }
      #pragma unroll
      for (int j = 0; j < 4; ++j) o[j][r] *= alpha;
    }
    __syncthreads();

    // P as two A-operand tiles (16x32 each) from LDS
    v16h pa[2];
    #pragma unroll
    for (int ks = 0; ks < 2; ++ks) pa[ks] = loadA(pl + mrow * 64 + ks * 32, half);

    // row sums via ones-WMMA: every lane gets its rows' sums in rs[r]
    v8f rs = {};
    rs = wmma_f16(pa[0], onesB, rs);
    rs = wmma_f16(pa[1], onesB, rs);

    // O += P @ V from transposed V (t-contiguous -> b128 loads)
    #pragma unroll
    for (int ks = 0; ks < 2; ++ks) {
      #pragma unroll
      for (int j = 0; j < 4; ++j) {
        const v16h vb = loadB(
            vT16 + vbase + (size_t)(j * 16 + mrow) * ROWS + t0 + ks * 32, half);
        o[j] = wmma_f16(pa[ks], vb, o[j]);
      }
    }

    #pragma unroll
    for (int r = 0; r < 8; ++r) lrun[r] += rs[r];
    __syncthreads();
  }

  // normalize and store fp16 at (b, s, h, d) row-major over 1024 cols
  #pragma unroll
  for (int j = 0; j < 4; ++j) {
    #pragma unroll
    for (int r = 0; r < 8; ++r) {
      const int row = qb + r + 8 * half;
      const float val = o[j][r] / lrun[r];
      ao16[base + (size_t)row * DM + j * 16 + mrow] = (_Float16)val;
    }
  }
}

// -------------------------------------------------------------------------
extern "C" void kernel_launch(void* const* d_in, const int* in_sizes, int n_in,
                              void* d_out, int out_size, void* d_ws, size_t ws_size,
                              hipStream_t stream) {
  const float* x    = (const float*)d_in[0];
  const float* ln_g = (const float*)d_in[1];
  const float* ln_b = (const float*)d_in[2];
  const float* wq   = (const float*)d_in[3];
  const float* bq   = (const float*)d_in[4];
  const float* wk   = (const float*)d_in[5];
  const float* bk   = (const float*)d_in[6];
  const float* wv   = (const float*)d_in[7];
  const float* bv   = (const float*)d_in[8];
  const float* wo   = (const float*)d_in[9];
  const float* bo   = (const float*)d_in[10];
  float* out = (float*)d_out;

  // workspace carve-up (fp16 tensors)
  const size_t actB = (size_t)ROWS * DM * sizeof(_Float16);   // 16 MB
  const size_t wB   = (size_t)DM * DM * sizeof(_Float16);     // 2 MB
  char* ws = (char*)d_ws;
  _Float16* x16   = (_Float16*)(ws);                 ws += actB;
  _Float16* xn16  = (_Float16*)(ws);                 ws += actB;
  _Float16* q16   = (_Float16*)(ws);                 ws += actB;
  _Float16* k16   = (_Float16*)(ws);                 ws += actB;
  _Float16* vT16  = (_Float16*)(ws);                 ws += actB;  // [n][m] transposed
  _Float16* ao16  = (_Float16*)(ws);                 ws += actB;
  _Float16* wq16T = (_Float16*)(ws);                 ws += wB;
  _Float16* wk16T = (_Float16*)(ws);                 ws += wB;
  _Float16* wv16T = (_Float16*)(ws);                 ws += wB;
  _Float16* wo16T = (_Float16*)(ws);                 ws += wB;

  // 1) LayerNorm + fp16 casts
  ln_cast_kernel<<<ROWS, 256, 0, stream>>>(x, ln_g, ln_b, x16, xn16);

  // 2) weight casts (transposed)
  const int wblocks = (DM * DM) / 256;
  cast_wT_kernel<<<wblocks, 256, 0, stream>>>(wq, wq16T);
  cast_wT_kernel<<<wblocks, 256, 0, stream>>>(wk, wk16T);
  cast_wT_kernel<<<wblocks, 256, 0, stream>>>(wv, wv16T);
  cast_wT_kernel<<<wblocks, 256, 0, stream>>>(wo, wo16T);

  // 3) Q/K/V projections (fp16 out; V stored transposed [feature][token])
  dim3 ggrid(ROWS / 32, DM / 64);
  gemm_wmma_kernel<0><<<ggrid, 32, 0, stream>>>(xn16, wq16T, bq, q16, nullptr);
  gemm_wmma_kernel<0><<<ggrid, 32, 0, stream>>>(x16,  wk16T, bk, k16, nullptr);
  gemm_wmma_kernel<1><<<ggrid, 32, 0, stream>>>(x16,  wv16T, bv, vT16, nullptr);

  // 4) attention
  attn_wmma_kernel<<<dim3(SEQ / 16, NH, NB), 32, 0, stream>>>(q16, k16, vT16, ao16);

  // 5) output projection (fp32 out + bias)
  gemm_wmma_kernel<2><<<ggrid, 32, 0, stream>>>(ao16, wo16T, bo, nullptr, out);
}
